// QuantumNATHybrid_65481071399602
// MI455X (gfx1250) — matrix operand order repaired and checked by reading, not compiled, not run on hardware
//
#include <hip/hip_runtime.h>

#define GAMMA  0.001f
#define NIMG   4096
#define MSUP   8192
#define KDIM   784
#define KPAD   800      // rows zero-padded to multiple of KC -> no tail guards
#define NCLASS 4

#define TN 128   // feats rows per workgroup (4 wave-rows x 32)
#define TM 64    // support rows per workgroup (2 wave-cols x 32)
#define KC 32    // K per WMMA chunk

typedef __attribute__((ext_vector_type(16))) __bf16          v16bf;
typedef __attribute__((ext_vector_type(8)))  unsigned short  v8u;
typedef __attribute__((ext_vector_type(8)))  float           v8f;

__device__ __forceinline__ unsigned short f2bf(float f) {
    unsigned int u = __float_as_uint(f);
    unsigned int r = u + 0x7FFFu + ((u >> 16) & 1u);   // round-to-nearest-even
    return (unsigned short)(r >> 16);
}
__device__ __forceinline__ float bf2f(unsigned short b) {
    return __uint_as_float(((unsigned int)b) << 16);
}

// load a 16-element bf16 fragment as two contiguous 16B global loads
__device__ __forceinline__ v16bf ldfrag(const unsigned short* __restrict__ p, int second) {
    v8u x0 = *(const v8u*)p;
    v8u x1 = *(const v8u*)(p + second);
    return __builtin_bit_cast(v16bf, __builtin_shufflevector(
        x0, x1, 0,1,2,3,4,5,6,7,8,9,10,11,12,13,14,15));
}

// ---------------------------------------------------------------- zero output
__global__ void zero_f32(float* __restrict__ p, int n) {
    int i = blockIdx.x * blockDim.x + threadIdx.x;
    if (i < n) p[i] = 0.0f;
}

// ------------------------------------------------- conv1 (1->8) + relu + pool
__global__ void conv1_pool(const float* __restrict__ x, const float* __restrict__ w1,
                           const float* __restrict__ b1, float* __restrict__ h1) {
    int idx = blockIdx.x * blockDim.x + threadIdx.x;
    if (idx >= NIMG * 8 * 14 * 14) return;
    int px = idx % 14; int t = idx / 14;
    int py = t % 14;   t /= 14;
    int c  = t % 8;    int n = t / 8;

    const float* xi = x + (size_t)n * 28 * 28;
    const float* w  = w1 + c * 9;
    float bias = b1[c];
    float m = 0.0f;                     // relu >= 0, so max starts at 0
    #pragma unroll
    for (int dy = 0; dy < 2; ++dy)
    #pragma unroll
    for (int dx = 0; dx < 2; ++dx) {
        int oy = py * 2 + dy, ox = px * 2 + dx;
        float s = bias;
        #pragma unroll
        for (int ky = 0; ky < 3; ++ky) {
            int iy = oy + ky - 1;
            if (iy < 0 || iy >= 28) continue;
            #pragma unroll
            for (int kx = 0; kx < 3; ++kx) {
                int ix = ox + kx - 1;
                if (ix < 0 || ix >= 28) continue;
                s = fmaf(xi[iy * 28 + ix], w[ky * 3 + kx], s);
            }
        }
        m = fmaxf(m, s);
    }
    h1[idx] = m;
}

// ----------------------------------------------- conv2 (8->16) + relu + pool
__global__ void conv2_pool(const float* __restrict__ h1, const float* __restrict__ w2,
                           const float* __restrict__ b2, float* __restrict__ feats) {
    int idx = blockIdx.x * blockDim.x + threadIdx.x;
    if (idx >= NIMG * 16 * 7 * 7) return;
    int px = idx % 7; int t = idx / 7;
    int py = t % 7;   t /= 7;
    int c  = t % 16;  int n = t / 16;

    float m = 0.0f;
    #pragma unroll
    for (int dy = 0; dy < 2; ++dy)
    #pragma unroll
    for (int dx = 0; dx < 2; ++dx) {
        int oy = py * 2 + dy, ox = px * 2 + dx;
        float s = b2[c];
        for (int ci = 0; ci < 8; ++ci) {
            const float* hp = h1 + (((size_t)n * 8 + ci) * 14) * 14;
            const float* w  = w2 + ((c * 8 + ci) * 9);
            #pragma unroll
            for (int ky = 0; ky < 3; ++ky) {
                int iy = oy + ky - 1;
                if (iy < 0 || iy >= 14) continue;
                #pragma unroll
                for (int kx = 0; kx < 3; ++kx) {
                    int ix = ox + kx - 1;
                    if (ix < 0 || ix >= 14) continue;
                    s = fmaf(hp[iy * 14 + ix], w[ky * 3 + kx], s);
                }
            }
        }
        m = fmaxf(m, s);
    }
    // reference flatten order: [c][y][x]
    feats[(size_t)n * KDIM + c * 49 + py * 7 + px] = m;
}

// -------------------------------------------------- per-row sum of squares
__global__ void row_sumsq(const float* __restrict__ v, float* __restrict__ out, int ncols) {
    __shared__ float red[256];
    int row = blockIdx.x;
    const float* p = v + (size_t)row * ncols;
    float s = 0.0f;
    for (int k = threadIdx.x; k < ncols; k += blockDim.x) {
        float a = p[k];
        s = fmaf(a, a, s);
    }
    red[threadIdx.x] = s;
    __syncthreads();
    for (int off = 128; off > 0; off >>= 1) {
        if ((int)threadIdx.x < off) red[threadIdx.x] += red[threadIdx.x + off];
        __syncthreads();
    }
    if (threadIdx.x == 0) out[row] = red[0];
}

// --------------------------- split fp32 matrix into hi/lo bf16 planes (padded)
__global__ void split_bf16(const float* __restrict__ src, unsigned short* __restrict__ hi,
                           unsigned short* __restrict__ lo, int nrows) {
    int idx = blockIdx.x * blockDim.x + threadIdx.x;
    if (idx >= nrows * KPAD) return;
    int row = idx / KPAD, k = idx - row * KPAD;
    float v = (k < KDIM) ? src[(size_t)row * KDIM + k] : 0.0f;
    unsigned short h = f2bf(v);
    hi[idx] = h;
    lo[idx] = f2bf(v - bf2f(h));
}

// ---------------------------------------------------------------------------
// Fused RBF kernel GEMM from pre-split bf16 planes (all L2-resident):
//   dot = feats.support^T via split-bf16 (hi*hi + hi*lo + lo*hi)
//   d2  = f2 + s2 - 2*dot ; K = exp(-g*d2) ; out += K @ alpha (atomics)
// No LDS, no barriers: every fragment is 2x global_load_b128 of a bf16 row.
// Each wave computes a 32x32 output tile (4 accumulators, 12 WMMA / chunk).
// K loop kept at unroll 1 so accumulators stay pinned (no pipeliner register
// rotation -> no v_mov_b64 copies + hazard NOP blocks between WMMAs).
// ---------------------------------------------------------------------------
__global__ __launch_bounds__(256) void rbf_gemm(
    const unsigned short* __restrict__ fH, const unsigned short* __restrict__ fL,
    const unsigned short* __restrict__ sH, const unsigned short* __restrict__ sL,
    const float* __restrict__ f2,  const float* __restrict__ s2v,
    const float* __restrict__ alpha, float* __restrict__ out) {

    const int tid    = threadIdx.x;
    const int lane   = tid & 31;
    const int wave   = tid >> 5;
    const int wr     = wave & 3;     // 4 wave-rows  -> 128 N
    const int wc     = wave >> 2;    // 2 wave-cols  -> 64 M
    const int bn     = blockIdx.y * TN;
    const int bm     = blockIdx.x * TM;
    const int halfId = lane >> 4;
    const int l16    = lane & 15;

    const int arow0 = bn + wr * 32 + l16;   // A rows this lane touches
    const int arow1 = arow0 + 16;
    const int bcol0 = bm + wc * 32 + l16;   // B (support) rows this lane touches
    const int bcol1 = bcol0 + 16;

    // row base pointers into the four bf16 planes
    const unsigned short* pAh0 = fH + (size_t)arow0 * KPAD;
    const unsigned short* pAl0 = fL + (size_t)arow0 * KPAD;
    const unsigned short* pAh1 = fH + (size_t)arow1 * KPAD;
    const unsigned short* pAl1 = fL + (size_t)arow1 * KPAD;
    const unsigned short* pBh0 = sH + (size_t)bcol0 * KPAD;
    const unsigned short* pBl0 = sL + (size_t)bcol0 * KPAD;
    const unsigned short* pBh1 = sH + (size_t)bcol1 * KPAD;
    const unsigned short* pBl1 = sL + (size_t)bcol1 * KPAD;

    const int aOff = 8 * halfId;    // A fragment: runs at k0+8h and k0+16+8h
    const int bOff = 8 * halfId;    // B fragment: runs at k0+8h and k0+16+8h

    v8f acc00 = {}, acc01 = {}, acc10 = {}, acc11 = {};

    #pragma unroll 1
    for (int k0 = 0; k0 < KPAD; k0 += KC) {
        v16bf a0h = ldfrag(pAh0 + k0 + aOff, 16);
        v16bf a0l = ldfrag(pAl0 + k0 + aOff, 16);
        v16bf a1h = ldfrag(pAh1 + k0 + aOff, 16);
        v16bf a1l = ldfrag(pAl1 + k0 + aOff, 16);
        v16bf b0h = ldfrag(pBh0 + k0 + bOff, 16);
        v16bf b0l = ldfrag(pBl0 + k0 + bOff, 16);
        v16bf b1h = ldfrag(pBh1 + k0 + bOff, 16);
        v16bf b1l = ldfrag(pBl1 + k0 + bOff, 16);

        acc00 = __builtin_amdgcn_wmma_f32_16x16x32_bf16(false, a0h, false, b0h, (short)0, acc00, false, false);
        acc01 = __builtin_amdgcn_wmma_f32_16x16x32_bf16(false, a0h, false, b1h, (short)0, acc01, false, false);
        acc10 = __builtin_amdgcn_wmma_f32_16x16x32_bf16(false, a1h, false, b0h, (short)0, acc10, false, false);
        acc11 = __builtin_amdgcn_wmma_f32_16x16x32_bf16(false, a1h, false, b1h, (short)0, acc11, false, false);

        acc00 = __builtin_amdgcn_wmma_f32_16x16x32_bf16(false, a0h, false, b0l, (short)0, acc00, false, false);
        acc01 = __builtin_amdgcn_wmma_f32_16x16x32_bf16(false, a0h, false, b1l, (short)0, acc01, false, false);
        acc10 = __builtin_amdgcn_wmma_f32_16x16x32_bf16(false, a1h, false, b0l, (short)0, acc10, false, false);
        acc11 = __builtin_amdgcn_wmma_f32_16x16x32_bf16(false, a1h, false, b1l, (short)0, acc11, false, false);

        acc00 = __builtin_amdgcn_wmma_f32_16x16x32_bf16(false, a0l, false, b0h, (short)0, acc00, false, false);
        acc01 = __builtin_amdgcn_wmma_f32_16x16x32_bf16(false, a0l, false, b1h, (short)0, acc01, false, false);
        acc10 = __builtin_amdgcn_wmma_f32_16x16x32_bf16(false, a1l, false, b0h, (short)0, acc10, false, false);
        acc11 = __builtin_amdgcn_wmma_f32_16x16x32_bf16(false, a1l, false, b1h, (short)0, acc11, false, false);
    }

    // ---------------- epilogue: RBF + alpha contraction, fused ----------------
    const float s2c0 = s2v[bcol0];
    const float s2c1 = s2v[bcol1];
    const float a00 = alpha[bcol0 * 4 + 0], a01 = alpha[bcol0 * 4 + 1];
    const float a02 = alpha[bcol0 * 4 + 2], a03 = alpha[bcol0 * 4 + 3];
    const float a10 = alpha[bcol1 * 4 + 0], a11 = alpha[bcol1 * 4 + 1];
    const float a12 = alpha[bcol1 * 4 + 2], a13 = alpha[bcol1 * 4 + 3];

    #pragma unroll
    for (int ti = 0; ti < 2; ++ti) {
        const v8f* accA = ti ? &acc10 : &acc00;   // column-tile 0
        const v8f* accB = ti ? &acc11 : &acc01;   // column-tile 1
        #pragma unroll
        for (int r = 0; r < 8; ++r) {
            int row  = bn + wr * 32 + ti * 16 + halfId * 8 + r;  // C layout: M = r + 8*(lane/16)
            float kv0 = __expf(-GAMMA * (f2[row] + s2c0 - 2.0f * (*accA)[r]));
            float kv1 = __expf(-GAMMA * (f2[row] + s2c1 - 2.0f * (*accB)[r]));
            float c0 = kv0 * a00 + kv1 * a10;
            float c1 = kv0 * a01 + kv1 * a11;
            float c2 = kv0 * a02 + kv1 * a12;
            float c3 = kv0 * a03 + kv1 * a13;
            #pragma unroll
            for (int off = 8; off > 0; off >>= 1) {   // reduce over the 16 columns
                c0 += __shfl_xor(c0, off, 16);
                c1 += __shfl_xor(c1, off, 16);
                c2 += __shfl_xor(c2, off, 16);
                c3 += __shfl_xor(c3, off, 16);
            }
            if (l16 == 0) {
                atomicAdd(&out[row * 4 + 0], c0);
                atomicAdd(&out[row * 4 + 1], c1);
                atomicAdd(&out[row * 4 + 2], c2);
                atomicAdd(&out[row * 4 + 3], c3);
            }
        }
    }
}

extern "C" void kernel_launch(void* const* d_in, const int* in_sizes, int n_in,
                              void* d_out, int out_size, void* d_ws, size_t ws_size,
                              hipStream_t stream) {
    const float* x       = (const float*)d_in[0];
    const float* w1      = (const float*)d_in[1];
    const float* b1      = (const float*)d_in[2];
    const float* w2      = (const float*)d_in[3];
    const float* b2      = (const float*)d_in[4];
    const float* support = (const float*)d_in[5];
    const float* alpha   = (const float*)d_in[6];
    float* out = (float*)d_out;

    // ---- workspace layout (bytes), all regions rewritten every launch ----
    char* ws = (char*)d_ws;
    const size_t h1Bytes    = (size_t)NIMG * 8 * 14 * 14 * 4;   // 25.69 MB
    const size_t featsBytes = (size_t)NIMG * KDIM * 4;          // 12.85 MB

    float* h1    = (float*)ws;                                   // dead after conv2
    unsigned short* fH = (unsigned short*)ws;                    // aliases h1 (fH+fL < h1)
    unsigned short* fL = fH + (size_t)NIMG * KPAD;
    float* feats = (float*)(ws + h1Bytes);
    float* f2    = (float*)(ws + h1Bytes + featsBytes);
    float* s2v   = f2 + NIMG;
    unsigned short* sH = (unsigned short*)(ws + h1Bytes + featsBytes + 49152);
    unsigned short* sL = sH + (size_t)MSUP * KPAD;

    zero_f32<<<(NIMG * NCLASS + 255) / 256, 256, 0, stream>>>(out, NIMG * NCLASS);

    conv1_pool<<<(NIMG * 8 * 14 * 14 + 255) / 256, 256, 0, stream>>>(x, w1, b1, h1);
    conv2_pool<<<(NIMG * 16 * 7 * 7 + 255) / 256, 256, 0, stream>>>(h1, w2, b2, feats);

    row_sumsq<<<NIMG, 256, 0, stream>>>(feats, f2, KDIM);
    row_sumsq<<<MSUP, 256, 0, stream>>>(support, s2v, KDIM);

    // h1 is dead now: split planes may overwrite it
    split_bf16<<<(NIMG * KPAD + 255) / 256, 256, 0, stream>>>(feats, fH, fL, NIMG);
    split_bf16<<<(MSUP * KPAD + 255) / 256, 256, 0, stream>>>(support, sH, sL, MSUP);

    dim3 grid(MSUP / TM, NIMG / TN);   // 128 x 32 workgroups
    rbf_gemm<<<grid, 256, 0, stream>>>(fH, fL, sH, sL, f2, s2v, alpha, out);
}
